// Upsample_223338299759
// MI455X (gfx1250) — matrix operand order repaired and checked by reading, not compiled, not run on hardware
//
#include <hip/hip_runtime.h>

// CARAFE upsample for MI455X (gfx1250, wave32, WMMA).
// Stage 1: 1x1 down conv as bf16 WMMA GEMM  -> t [N,H,W,64] fp32 in ws
// Stage 2: 3x3 encoder conv + softmax (VALU) -> kern [N,H,W,4,9] fp32 in ws
// Stage 3: fused reassembly + pixel_shuffle + 1x1 out conv (bf16 WMMA),
//          upsampled activations live only in LDS (never hit HBM).

#define CKS  3
#define UPF  2
#define NB   4
#define CIN  256
#define HIN  128
#define WIN  128
#define CD   64          // C/4
#define EC   36          // UP*UP*K*K
#define OUTC 256
#define H2   (HIN * UPF)
#define W2   (WIN * UPF)

typedef __bf16 bf16_t;
typedef bf16_t bf16x16 __attribute__((ext_vector_type(16)));
typedef bf16_t bf16x8  __attribute__((ext_vector_type(8)));
typedef float  f32x8   __attribute__((ext_vector_type(8)));

union Frag { bf16x16 v; bf16x8 h[2]; };

__device__ __forceinline__ unsigned short f2bf(float f) {
  union { float f; unsigned int u; } cv; cv.f = f;
  const unsigned int u = cv.u;
  return (unsigned short)((u + 0x7FFFu + ((u >> 16) & 1u)) >> 16); // RNE
}

// ---------------------------------------------------------------------------
// Kernel 1: t[pix][64] = w_down(64x256) @ x[:,pix] + b_down   (bf16 WMMA)
// block = 256 threads (8 waves), tile = 64 oc x 32 pixels
// ---------------------------------------------------------------------------
__global__ void down_wmma(const float* __restrict__ x,
                          const float* __restrict__ w_down,
                          const float* __restrict__ b_down,
                          float* __restrict__ tws) {
  extern __shared__ char smem[];
  unsigned short* lds_w = (unsigned short*)smem;               // [64][256] bf16
  unsigned short* lds_x = (unsigned short*)(smem + CD*CIN*2);  // [32][264] bf16

  const int t   = threadIdx.x;
  const int p0  = blockIdx.x * 32;          // 32 consecutive pixels, same row
  const int n   = p0 >> 14;                 // / (H*W)
  const int rem = p0 & 16383;
  const int h   = rem >> 7;
  const int w0  = rem & 127;

  for (int i = t; i < CD * CIN; i += 256) lds_w[i] = f2bf(w_down[i]);

  {
    const int pix = t & 31;
    const int cb  = t >> 5;
    for (int cc = 0; cc < CIN; cc += 8) {
      const int c = cc + cb;
      const float v = x[(((n * CIN + c) * HIN + h) * WIN) + w0 + pix];
      lds_x[pix * 264 + c] = f2bf(v);
    }
  }
  __syncthreads();

  const int wv   = t >> 5;
  const int lane = t & 31;
  const int oc_base  = (wv >> 1) * 16;
  const int pix_base = (wv & 1) * 16;
  const int hi = (lane & 16) ? 8 : 0;

  f32x8 acc;
  #pragma unroll
  for (int r = 0; r < 8; ++r) acc[r] = b_down[oc_base + r + hi];

  const int m    = oc_base  + (lane & 15);
  const int pixl = pix_base + (lane & 15);

  #pragma unroll
  for (int k0 = 0; k0 < CIN; k0 += 32) {
    Frag a, b;
    const unsigned short* ap = lds_w + m * 256 + k0 + ((lane & 16) ? 8 : 0);
    a.h[0] = *(const bf16x8*)ap;
    a.h[1] = *(const bf16x8*)(ap + 16);
    const unsigned short* bp = lds_x + pixl * 264 + k0 + ((lane & 16) ? 8 : 0);
    b.h[0] = *(const bf16x8*)bp;
    b.h[1] = *(const bf16x8*)(bp + 16);
    acc = __builtin_amdgcn_wmma_f32_16x16x32_bf16(false, a.v, false, b.v,
                                                  (short)0, acc, false, false);
  }

  const int pg = p0 + pixl;
  #pragma unroll
  for (int r = 0; r < 8; ++r)
    tws[pg * CD + oc_base + r + hi] = acc[r];
}

// ---------------------------------------------------------------------------
// Kernel 2: encoder 3x3 conv (64->36) + softmax over k^2 per sub-pixel.
// One thread per pixel; w_enc reordered [tap][c][36] in LDS (81 KB, CDNA5
// 320KB LDS). kern layout: [pix][q][9].
// ---------------------------------------------------------------------------
__device__ __forceinline__ void accum36(float tv, const float* __restrict__ wr,
                                        float (&acc)[EC]) {
  const float4* wp = (const float4*)wr;   // rows are 144B => 16B aligned
  #pragma unroll
  for (int o4 = 0; o4 < 9; ++o4) {
    const float4 wq = wp[o4];
    acc[o4 * 4 + 0] += tv * wq.x;
    acc[o4 * 4 + 1] += tv * wq.y;
    acc[o4 * 4 + 2] += tv * wq.z;
    acc[o4 * 4 + 3] += tv * wq.w;
  }
}

__global__ void enc_softmax(const float* __restrict__ tws,
                            const float* __restrict__ w_enc,
                            const float* __restrict__ b_enc,
                            float* __restrict__ kernws) {
  extern __shared__ char smem[];
  float* lds_we = (float*)smem;           // [9][64][36] fp32
  const int t = threadIdx.x;

  for (int i = t; i < EC * CD * 9; i += 256) {
    const int o   = i / (CD * 9);
    const int r   = i % (CD * 9);
    const int c   = r / 9;
    const int tap = r % 9;                // w_enc flat = (o*64+c)*9 + tap
    lds_we[(tap * CD + c) * EC + o] = w_enc[i];
  }
  __syncthreads();

  const int p   = blockIdx.x * 256 + t;
  const int n   = p >> 14;
  const int rem = p & 16383;
  const int h   = rem >> 7;
  const int w   = rem & 127;

  float acc[EC];
  #pragma unroll
  for (int o = 0; o < EC; ++o) acc[o] = b_enc[o];

  int tap = 0;
  for (int dh = -1; dh <= 1; ++dh) {
    const int hh = h + dh;
    for (int dw = -1; dw <= 1; ++dw, ++tap) {
      const int ww = w + dw;
      if (hh < 0 || hh >= HIN || ww < 0 || ww >= WIN) continue;  // zero pad
      const float* tb = tws + (((n * HIN + hh) * WIN) + ww) * CD;
      for (int c4 = 0; c4 < CD / 4; ++c4) {
        const float4 tv = *(const float4*)(tb + c4 * 4);
        const float* wr = lds_we + ((tap * CD + c4 * 4) * EC);
        accum36(tv.x, wr,          acc);
        accum36(tv.y, wr + EC,     acc);
        accum36(tv.z, wr + 2 * EC, acc);
        accum36(tv.w, wr + 3 * EC, acc);
      }
    }
  }

  // softmax over the 9 taps for each of the 4 sub-pixel positions (o = kk*4+q)
  float* kout = kernws + (size_t)p * EC;
  #pragma unroll
  for (int q = 0; q < 4; ++q) {
    float mx = acc[q];
    #pragma unroll
    for (int kk = 1; kk < 9; ++kk) mx = fmaxf(mx, acc[kk * 4 + q]);
    float e[9]; float s = 0.f;
    #pragma unroll
    for (int kk = 0; kk < 9; ++kk) { e[kk] = __expf(acc[kk * 4 + q] - mx); s += e[kk]; }
    const float inv = 1.0f / s;
    #pragma unroll
    for (int kk = 0; kk < 9; ++kk) kout[q * 9 + kk] = e[kk] * inv;
  }
}

// ---------------------------------------------------------------------------
// Kernel 3: fused reassembly + pixel_shuffle + 1x1 out conv.
// Block owns 32 upsampled pixels of one row; builds up[256c][32pix] in LDS
// (9 FMAs/value, x gathered from L2), stages all of w_out as bf16 in LDS
// (128 KB), then 256x256x32 GEMM via v_wmma_f32_16x16x32_bf16.
// ---------------------------------------------------------------------------
__global__ void carafe_out_wmma(const float* __restrict__ x,
                                const float* __restrict__ kernws,
                                const float* __restrict__ w_out,
                                const float* __restrict__ b_out,
                                float* __restrict__ out) {
  extern __shared__ char smem[];
  unsigned short* lds_A   = (unsigned short*)smem;                       // [256][256] bf16
  unsigned short* lds_up  = (unsigned short*)(smem + OUTC * CIN * 2);    // [32][264]  bf16
  float*          lds_kern = (float*)(smem + OUTC * CIN * 2 + 32 * 264 * 2); // [32][9]

  const int t   = threadIdx.x;
  const int p0  = blockIdx.x * 32;
  const int n   = p0 / (H2 * W2);
  const int rem = p0 % (H2 * W2);
  const int y   = rem / W2;
  const int x0  = rem % W2;                // multiple of 32
  const int hc  = y >> 1;
  const int iy  = y & 1;

  for (int i = t; i < OUTC * CIN; i += 256) lds_A[i] = f2bf(w_out[i]);

  for (int idx = t; idx < 32 * 9; idx += 256) {
    const int tp = idx / 9, kk = idx % 9;
    const int xx = x0 + tp;
    const int wc = xx >> 1;
    const int q  = (iy << 1) | (xx & 1);
    lds_kern[idx] = kernws[(size_t)((n * HIN + hc) * WIN + wc) * EC + q * 9 + kk];
  }
  __syncthreads();

  // ---- build the up[c][pix] tile in LDS (per-pixel 3x3 reassembly) ----
  {
    const int pix = t & 31;
    const int xx  = x0 + pix;
    const int wc  = xx >> 1;
    float kv[9];
    #pragma unroll
    for (int kk = 0; kk < 9; ++kk) kv[kk] = lds_kern[pix * 9 + kk];
    int  toff[9]; bool tok[9];
    {
      int kk = 0;
      #pragma unroll
      for (int dh = -1; dh <= 1; ++dh)
        #pragma unroll
        for (int dw = -1; dw <= 1; ++dw, ++kk) {
          const int hh = hc + dh, ww = wc + dw;
          tok[kk]  = (hh >= 0 && hh < HIN && ww >= 0 && ww < WIN);
          toff[kk] = hh * WIN + ww;
        }
    }
    for (int it = 0; it < 32; ++it) {
      const int c = (t >> 5) + it * 8;
      const float* xb = x + (size_t)(n * CIN + c) * (HIN * WIN);
      __builtin_prefetch(xb + toff[4], 0, 1);      // global_prefetch_b8
      float a = 0.f;
      #pragma unroll
      for (int kk = 0; kk < 9; ++kk)
        if (tok[kk]) a += kv[kk] * xb[toff[kk]];
      lds_up[pix * 264 + c] = f2bf(a);
    }
  }
  __syncthreads();

  // ---- 256(oc) x 32(pix) GEMM over K=256, 8 waves * 2 M-tiles * 2 N-tiles ----
  const int wv   = t >> 5;
  const int lane = t & 31;
  const int hi   = (lane & 16) ? 8 : 0;

  f32x8 acc[2][2];
  #pragma unroll
  for (int m2 = 0; m2 < 2; ++m2) {
    const int ocb = (wv * 2 + m2) * 16;
    #pragma unroll
    for (int r = 0; r < 8; ++r) {
      const float bb = b_out[ocb + r + hi];
      acc[m2][0][r] = bb;
      acc[m2][1][r] = bb;
    }
  }

  const int mrow0 = (wv * 2 + 0) * 16 + (lane & 15);
  const int mrow1 = (wv * 2 + 1) * 16 + (lane & 15);
  const int pc0   = 0 * 16 + (lane & 15);
  const int pc1   = 1 * 16 + (lane & 15);

  #pragma unroll
  for (int k0 = 0; k0 < CIN; k0 += 32) {
    Frag a0, a1, b0, b1;
    const unsigned short* ap0 = lds_A + mrow0 * 256 + k0 + ((lane & 16) ? 8 : 0);
    a0.h[0] = *(const bf16x8*)ap0; a0.h[1] = *(const bf16x8*)(ap0 + 16);
    const unsigned short* ap1 = lds_A + mrow1 * 256 + k0 + ((lane & 16) ? 8 : 0);
    a1.h[0] = *(const bf16x8*)ap1; a1.h[1] = *(const bf16x8*)(ap1 + 16);
    const unsigned short* bp0 = lds_up + pc0 * 264 + k0 + ((lane & 16) ? 8 : 0);
    b0.h[0] = *(const bf16x8*)bp0; b0.h[1] = *(const bf16x8*)(bp0 + 16);
    const unsigned short* bp1 = lds_up + pc1 * 264 + k0 + ((lane & 16) ? 8 : 0);
    b1.h[0] = *(const bf16x8*)bp1; b1.h[1] = *(const bf16x8*)(bp1 + 16);
    acc[0][0] = __builtin_amdgcn_wmma_f32_16x16x32_bf16(false, a0.v, false, b0.v, (short)0, acc[0][0], false, false);
    acc[0][1] = __builtin_amdgcn_wmma_f32_16x16x32_bf16(false, a0.v, false, b1.v, (short)0, acc[0][1], false, false);
    acc[1][0] = __builtin_amdgcn_wmma_f32_16x16x32_bf16(false, a1.v, false, b0.v, (short)0, acc[1][0], false, false);
    acc[1][1] = __builtin_amdgcn_wmma_f32_16x16x32_bf16(false, a1.v, false, b1.v, (short)0, acc[1][1], false, false);
  }

  #pragma unroll
  for (int m2 = 0; m2 < 2; ++m2) {
    const int ocb = (wv * 2 + m2) * 16 + hi;
    #pragma unroll
    for (int n2 = 0; n2 < 2; ++n2) {
      const int xx = x0 + n2 * 16 + (lane & 15);
      #pragma unroll
      for (int r = 0; r < 8; ++r)
        out[(((size_t)n * OUTC + ocb + r) * H2 + y) * W2 + xx] = acc[m2][n2][r];
    }
  }
}

// ---------------------------------------------------------------------------
extern "C" void kernel_launch(void* const* d_in, const int* in_sizes, int n_in,
                              void* d_out, int out_size, void* d_ws, size_t ws_size,
                              hipStream_t stream) {
  const float* x      = (const float*)d_in[0];
  const float* w_down = (const float*)d_in[1];
  const float* b_down = (const float*)d_in[2];
  const float* w_enc  = (const float*)d_in[3];
  const float* b_enc  = (const float*)d_in[4];
  const float* w_out  = (const float*)d_in[5];
  const float* b_out  = (const float*)d_in[6];
  float* out = (float*)d_out;

  float* tws    = (float*)d_ws;                       // N*H*W*64  fp32 (16 MB)
  float* kernws = tws + (size_t)NB * HIN * WIN * CD;  // N*H*W*36  fp32 (9.4 MB)

  const size_t sh1 = (size_t)CD * CIN * 2 + 32 * 264 * 2;               // ~49.7 KB
  down_wmma<<<(NB * HIN * WIN) / 32, 256, sh1, stream>>>(x, w_down, b_down, tws);

  const size_t sh2 = (size_t)9 * CD * EC * 4;                           // ~83 KB
  enc_softmax<<<(NB * HIN * WIN) / 256, 256, sh2, stream>>>(tws, w_enc, b_enc, kernws);

  const size_t sh3 = (size_t)OUTC * CIN * 2 + 32 * 264 * 2 + 32 * 9 * 4; // ~149 KB
  carafe_out_wmma<<<(NB * H2 * W2) / 32, 256, sh3, stream>>>(x, kernws, w_out, b_out, out);
}
